// GGNNBlockFeats_7842610283353
// MI455X (gfx1250) — matrix operand order repaired and checked by reading, not compiled, not run on hardware
//
#include <hip/hip_runtime.h>
#include <hip/hip_bf16.h>
#include <math.h>

#define NNODES 50000
#define CH     256
#define CH3    768
#define NEDGES 1600000
#define NSTEPS 3
#define NTYPES 3
#define NC     (NNODES * CH)

typedef __attribute__((ext_vector_type(16))) __bf16 v16bf;
typedef __attribute__((ext_vector_type(8)))  float  v8f;

__device__ __forceinline__ unsigned short f2bf(float f) {
    unsigned int u = __float_as_uint(f);
    u += 0x7FFFu + ((u >> 16) & 1u);      // round-to-nearest-even
    return (unsigned short)(u >> 16);
}

union BfFrag { v16bf v; uint4 q[2]; unsigned short u[16]; };

// A fragment: 16x32 bf16 (MxK). lane&15 = row M; per CDNA5 ISA:
// lanes 0-15: K = kb+0..7 then kb+16..23 ; lanes 16-31: K = kb+8..15 then kb+24..31
__device__ __forceinline__ v16bf load_a_frag(const unsigned short* __restrict__ base,
                                             int stride, int kb, int lane) {
    BfFrag f;
    const int half = lane >> 4;
    const unsigned short* p = base + (size_t)(lane & 15) * stride + kb + 8 * half;
    f.q[0] = *(const uint4*)(p);        // 8 bf16
    f.q[1] = *(const uint4*)(p + 16);   // 8 bf16, K + 16
    return f.v;
}

// Same A layout but source is fp32; pack to bf16 in-register (co-executes with WMMA).
__device__ __forceinline__ v16bf load_a_frag_f32(const float* __restrict__ base,
                                                 int stride, int kb, int lane) {
    const int half = lane >> 4;
    const float* p = base + (size_t)(lane & 15) * stride + kb + 8 * half;
    float4 f0 = *(const float4*)(p);
    float4 f1 = *(const float4*)(p + 4);
    float4 f2 = *(const float4*)(p + 16);
    float4 f3 = *(const float4*)(p + 20);
    BfFrag r;
    r.u[0]  = f2bf(f0.x); r.u[1]  = f2bf(f0.y); r.u[2]  = f2bf(f0.z); r.u[3]  = f2bf(f0.w);
    r.u[4]  = f2bf(f1.x); r.u[5]  = f2bf(f1.y); r.u[6]  = f2bf(f1.z); r.u[7]  = f2bf(f1.w);
    r.u[8]  = f2bf(f2.x); r.u[9]  = f2bf(f2.y); r.u[10] = f2bf(f2.z); r.u[11] = f2bf(f2.w);
    r.u[12] = f2bf(f3.x); r.u[13] = f2bf(f3.y); r.u[14] = f2bf(f3.z); r.u[15] = f2bf(f3.w);
    return r.v;
}

// B fragment: 32x16 bf16 (KxN), sourced from Wt[n][k] (weights stored out-major).
// lane&15 = column N; same K order as A: K = kb+8*half+0..7 then kb+16+8*half+0..7
__device__ __forceinline__ v16bf load_b_frag(const unsigned short* __restrict__ base,
                                             int stride, int kb, int lane) {
    BfFrag f;
    const int half = lane >> 4;
    const unsigned short* p = base + (size_t)(lane & 15) * stride + kb + 8 * half;
    f.q[0] = *(const uint4*)(p);
    f.q[1] = *(const uint4*)(p + 16);
    return f.v;
}

__device__ __forceinline__ v8f wmma_bf16(v16bf a, v16bf b, v8f c) {
    return __builtin_amdgcn_wmma_f32_16x16x32_bf16(false, a, false, b, (short)0, c,
                                                   false, false);
}

// ---------------- small utility kernels ----------------

__global__ void fill0_kernel(float* __restrict__ p, int n) {
    int i = blockIdx.x * blockDim.x + threadIdx.x;
    if (i < n) p[i] = 0.0f;
}

__global__ void cvt_bf16_kernel(const float* __restrict__ s, unsigned short* __restrict__ d, int n) {
    int i = blockIdx.x * blockDim.x + threadIdx.x;
    if (i < n) d[i] = f2bf(s[i]);
}

__global__ void init_h_kernel(const float* __restrict__ hin, float* __restrict__ hw,
                              unsigned short* __restrict__ hb, int n) {
    int i = blockIdx.x * blockDim.x + threadIdx.x;
    if (i < n) { float v = hin[i]; hw[i] = v; hb[i] = f2bf(v); }
}

// conv_w[ts][k][co] (fp32) -> Wt[ts][co][k] (bf16), ts in [0,9)
__global__ void cvt_convw_kernel(const float* __restrict__ W, unsigned short* __restrict__ Wt,
                                 int total) {
    int i = blockIdx.x * blockDim.x + threadIdx.x;
    if (i >= total) return;
    int ts  = i / (CH * CH);
    int rem = i - ts * (CH * CH);
    int co  = rem / CH;
    int k   = rem - co * CH;
    Wt[i] = f2bf(W[(size_t)ts * CH * CH + (size_t)k * CH + co]);
}

// One-time per-type edge compaction: int64 (src,dst,type) -> int32 (src,dst) lists.
__global__ void compact_edges_kernel(const long long* __restrict__ src,
                                     const long long* __restrict__ dst,
                                     const long long* __restrict__ et,
                                     int* __restrict__ cnt,       // [NTYPES]
                                     int2* __restrict__ elist,    // [NTYPES][NEDGES]
                                     int E) {
    int e = blockIdx.x * blockDim.x + threadIdx.x;
    if (e >= E) return;
    int t = (int)et[e];
    int pos = atomicAdd(&cnt[t], 1);
    elist[(size_t)t * NEDGES + pos] = make_int2((int)src[e], (int)dst[e]);
}

// ---------------- conv GEMM: m[n, co] = sum_k hb[n,k] * Wt[co,k] ----------------

__global__ __launch_bounds__(256)
void gemm_nt_wmma(const unsigned short* __restrict__ A,   // [N, 256] bf16
                  const unsigned short* __restrict__ W,   // [M, 256] bf16 (out-major)
                  float* __restrict__ Out, int M) {
    const int wave = threadIdx.x >> 5;
    const int lane = threadIdx.x & 31;
    const int colBase = (blockIdx.x * 8 + wave) * 16;
    const int rowBase = blockIdx.y * 16;
    const unsigned short* Arow = A + (size_t)rowBase * CH;
    const unsigned short* Wrow = W + (size_t)colBase * CH;

    v8f acc = {};
#pragma unroll
    for (int kb = 0; kb < CH; kb += 32) {
        v16bf a = load_a_frag(Arow, CH, kb, lane);
        v16bf b = load_b_frag(Wrow, CH, kb, lane);
        acc = wmma_bf16(a, b, acc);
    }
    const int n = lane & 15, half = lane >> 4;
    float* o = Out + (size_t)(rowBase + half * 8) * M + colBase + n;
#pragma unroll
    for (int v = 0; v < 8; ++v) o[(size_t)v * M] = acc[v];
}

// ---------------- compacted edge scatter: agg[dst] += m[src] ----------------

__global__ __launch_bounds__(256)
void scatter_compact_kernel(const float* __restrict__ m,
                            const int2* __restrict__ elist,
                            const int* __restrict__ cnt,   // points at cnt[t]
                            float* __restrict__ agg) {
    int e = blockIdx.x * 4 + (threadIdx.x >> 6);
    if (e >= cnt[0]) return;                 // uniform scalar load + exit
    int2 sd = elist[e];
    int c0 = (threadIdx.x & 63) << 2;
    const float4 val = *(const float4*)(m + (size_t)sd.x * CH + c0);
    float* ap = agg + (size_t)sd.y * CH + c0;
    atomicAdd(ap + 0, val.x);
    atomicAdd(ap + 1, val.y);
    atomicAdd(ap + 2, val.z);
    atomicAdd(ap + 3, val.w);
}

// ---------------- fused GRU: 6 WMMA accumulators + gates + state update ----------------

__global__ __launch_bounds__(256)
void gru_fused_wmma(const float* __restrict__ Agg,         // [N,256] fp32 (packed in-reg)
                    const unsigned short* __restrict__ HB, // [N,256] bf16
                    const unsigned short* __restrict__ Wih,// [768,256] bf16
                    const unsigned short* __restrict__ Whh,// [768,256] bf16
                    const float* __restrict__ bih,         // [768]
                    const float* __restrict__ bhh,         // [768]
                    const float* __restrict__ Hold,        // [N,256] fp32
                    float* __restrict__ Hnew,              // [N,256] fp32
                    unsigned short* __restrict__ HnewB,    // [N,256] bf16
                    float* __restrict__ HmsgAcc,           // [N,256] fp32
                    int lastStep) {
    const int wave = threadIdx.x >> 5;
    const int lane = threadIdx.x & 31;
    const int colBase = (blockIdx.x * 8 + wave) * 16;   // within [0,256)
    const int rowBase = blockIdx.y * 16;

    const float*          Ag = Agg + (size_t)rowBase * CH;
    const unsigned short* Hb = HB  + (size_t)rowBase * CH;
    const unsigned short* Wir = Wih + (size_t)(colBase) * CH;
    const unsigned short* Wiz = Wih + (size_t)(CH + colBase) * CH;
    const unsigned short* Win = Wih + (size_t)(2 * CH + colBase) * CH;
    const unsigned short* Whr = Whh + (size_t)(colBase) * CH;
    const unsigned short* Whz = Whh + (size_t)(CH + colBase) * CH;
    const unsigned short* Whn = Whh + (size_t)(2 * CH + colBase) * CH;

    v8f ir = {}, iz = {}, in_ = {}, hr = {}, hz = {}, hn = {};
#pragma unroll
    for (int kb = 0; kb < CH; kb += 32) {
        v16bf aa = load_a_frag_f32(Ag, CH, kb, lane);
        v16bf ah = load_a_frag(Hb, CH, kb, lane);
        ir  = wmma_bf16(aa, load_b_frag(Wir, CH, kb, lane), ir);
        iz  = wmma_bf16(aa, load_b_frag(Wiz, CH, kb, lane), iz);
        in_ = wmma_bf16(aa, load_b_frag(Win, CH, kb, lane), in_);
        hr  = wmma_bf16(ah, load_b_frag(Whr, CH, kb, lane), hr);
        hz  = wmma_bf16(ah, load_b_frag(Whz, CH, kb, lane), hz);
        hn  = wmma_bf16(ah, load_b_frag(Whn, CH, kb, lane), hn);
    }

    const int n = lane & 15, half = lane >> 4;
    const int col = colBase + n;
    const float b_ir = bih[col], b_iz = bih[CH + col], b_in = bih[2 * CH + col];
    const float b_hr = bhh[col], b_hz = bhh[CH + col], b_hn = bhh[2 * CH + col];
    const size_t rbase = (size_t)(rowBase + half * 8) * CH + col;
#pragma unroll
    for (int v = 0; v < 8; ++v) {
        size_t idx = rbase + (size_t)v * CH;
        float r  = 1.0f / (1.0f + __expf(-(ir[v] + b_ir + hr[v] + b_hr)));
        float z  = 1.0f / (1.0f + __expf(-(iz[v] + b_iz + hz[v] + b_hz)));
        float nn = tanhf(in_[v] + b_in + r * (hn[v] + b_hn));
        float ho = Hold[idx];
        float hv = (1.0f - z) * nn + z * ho;
        if (lastStep) {
            HmsgAcc[idx] += hv;            // fused h_msg accumulation
        } else {
            Hnew[idx]  = hv;
            HnewB[idx] = f2bf(hv);
        }
    }
}

// ---------------- final residual + LayerNorm + LeakyReLU ----------------

__global__ __launch_bounds__(256)
void final_ln_kernel(const float* __restrict__ Hin, const float* __restrict__ Hmsg,
                     const float* __restrict__ gamma, const float* __restrict__ beta,
                     float* __restrict__ Out) {
    __shared__ float sdata[256];
    const int row = blockIdx.x;
    const int tid = threadIdx.x;
    const size_t idx = (size_t)row * CH + tid;
    float x = Hin[idx] + Hmsg[idx];
    sdata[tid] = x;
    __syncthreads();
    for (int s = 128; s > 0; s >>= 1) {
        if (tid < s) sdata[tid] += sdata[tid + s];
        __syncthreads();
    }
    float mu = sdata[0] * (1.0f / CH);
    __syncthreads();
    float d = x - mu;
    sdata[tid] = d * d;
    __syncthreads();
    for (int s = 128; s > 0; s >>= 1) {
        if (tid < s) sdata[tid] += sdata[tid + s];
        __syncthreads();
    }
    float var = sdata[0] * (1.0f / CH);
    float y = d * rsqrtf(var + 1e-5f) * gamma[tid] + beta[tid];
    Out[idx] = (y >= 0.0f) ? y : 0.1f * y;
}

// ---------------- host orchestration ----------------

extern "C" void kernel_launch(void* const* d_in, const int* in_sizes, int n_in,
                              void* d_out, int out_size, void* d_ws, size_t ws_size,
                              hipStream_t stream) {
    const float*     h_in   = (const float*)d_in[0];
    const long long* eidx   = (const long long*)d_in[1];
    const long long* etype  = (const long long*)d_in[2];
    const float*     conv_w = (const float*)d_in[3];
    const float*     wih    = (const float*)d_in[4];
    const float*     whh    = (const float*)d_in[5];
    const float*     bih    = (const float*)d_in[6];
    const float*     bhh    = (const float*)d_in[7];
    const float*     gamma  = (const float*)d_in[8];
    const float*     beta   = (const float*)d_in[9];
    float*           out    = (float*)d_out;

    const long long* src = eidx;
    const long long* dst = eidx + NEDGES;

    char* ws = (char*)d_ws;
    size_t off = 0;
    auto alloc = [&](size_t bytes) -> char* {
        char* p = ws + off;
        off += (bytes + 255) & ~(size_t)255;
        return p;
    };

    float*          h0w    = (float*)alloc((size_t)NC * 4);
    unsigned short* h0b    = (unsigned short*)alloc((size_t)NC * 2);
    float*          hwA    = (float*)alloc((size_t)NC * 4);
    float*          hwB    = (float*)alloc((size_t)NC * 4);
    unsigned short* hbA    = (unsigned short*)alloc((size_t)NC * 2);
    unsigned short* hbB    = (unsigned short*)alloc((size_t)NC * 2);
    float*          mbuf   = (float*)alloc((size_t)NC * 4);
    float*          agg    = (float*)alloc((size_t)NC * 4);
    float*          hmsg   = (float*)alloc((size_t)NC * 4);
    int2*           elist  = (int2*)alloc((size_t)NTYPES * NEDGES * sizeof(int2));
    int*            cnt    = (int*)alloc(256);
    unsigned short* convwt = (unsigned short*)alloc((size_t)NTYPES * NSTEPS * CH * CH * 2);
    unsigned short* wihb   = (unsigned short*)alloc((size_t)NTYPES * CH3 * CH * 2);
    unsigned short* whhb   = (unsigned short*)alloc((size_t)NTYPES * CH3 * CH * 2);

    const int TPB = 256;
    const int gNC = (NC + TPB - 1) / TPB;
    const int wtot = NTYPES * NSTEPS * CH * CH;       // 589824
    const int gtot = NTYPES * CH3 * CH;               // 589824

    // one-time weight prep + edge compaction (deterministic per call)
    cvt_convw_kernel<<<(wtot + TPB - 1) / TPB, TPB, 0, stream>>>(conv_w, convwt, wtot);
    cvt_bf16_kernel<<<(gtot + TPB - 1) / TPB, TPB, 0, stream>>>(wih, wihb, gtot);
    cvt_bf16_kernel<<<(gtot + TPB - 1) / TPB, TPB, 0, stream>>>(whh, whhb, gtot);
    fill0_kernel<<<1, 64, 0, stream>>>((float*)cnt, NTYPES);   // 0.0f bits == 0
    compact_edges_kernel<<<(NEDGES + TPB - 1) / TPB, TPB, 0, stream>>>(
        src, dst, etype, cnt, elist, NEDGES);

    fill0_kernel<<<gNC, TPB, 0, stream>>>(hmsg, NC);
    init_h_kernel<<<gNC, TPB, 0, stream>>>(h_in, h0w, h0b, NC);

    const dim3 gemmGrid(CH / 128, NNODES / 16);       // (2, 3125)
    const dim3 gruGrid(CH / 128, NNODES / 16);        // (2, 3125)
    const int scatterBlocks = (NEDGES + 3) / 4;       // device-side count bound

    for (int t = 0; t < NTYPES; ++t) {
        float*          hwCur = h0w;  float*          hwNxt = hwA;
        unsigned short* hbCur = h0b;  unsigned short* hbNxt = hbA;

        for (int s = 0; s < NSTEPS; ++s) {
            fill0_kernel<<<gNC, TPB, 0, stream>>>(agg, NC);
            gemm_nt_wmma<<<gemmGrid, TPB, 0, stream>>>(
                hbCur, convwt + (size_t)(t * NSTEPS + s) * CH * CH, mbuf, CH);
            scatter_compact_kernel<<<scatterBlocks, TPB, 0, stream>>>(
                mbuf, elist + (size_t)t * NEDGES, cnt + t, agg);
            gru_fused_wmma<<<gruGrid, TPB, 0, stream>>>(
                agg, hbCur,
                wihb + (size_t)t * CH3 * CH, whhb + (size_t)t * CH3 * CH,
                bih + (size_t)t * CH3, bhh + (size_t)t * CH3,
                hwCur, hwNxt, hbNxt, hmsg, (s == NSTEPS - 1) ? 1 : 0);
            // advance ping-pong: h0 -> A -> B (last step writes only hmsg)
            hwCur = hwNxt; hbCur = hbNxt;
            hwNxt = (hwCur == hwA) ? hwB : hwA;
            hbNxt = (hbCur == hbA) ? hbB : hbA;
        }
    }

    final_ln_kernel<<<NNODES, TPB, 0, stream>>>(h_in, hmsg, gamma, beta, out);
}
